// Linear_6373731467798
// MI455X (gfx1250) — 8-bit matrix operand order repaired and checked by reading, compiled, not run on hardware
//
#include <hip/hip_runtime.h>


#define NROWS 32768
#define DIN   1024
#define DOUT  1024
#define MB    64
#define LDS_STRIDE 1096

typedef __attribute__((ext_vector_type(8))) int v8i;

__device__ __forceinline__ int sgn_i(float v) {
    return (v > 0.0f) - (v < 0.0f);
}

__global__ void hb_quant_w_packed(const float* __restrict__ w,
                                  char* __restrict__ pw8,
                                  float* __restrict__ sum_abs) {
    __shared__ float red[256];
    const long idx4 = (long)blockIdx.x * blockDim.x + threadIdx.x;
    const long i0   = idx4 * 4;
    const int  o    = (int)(i0 >> 10);
    const int  k    = (int)(i0 & 1023);
    const float4 wv = *(const float4*)(w + i0);
    const int s0 = sgn_i(wv.x);
    const int s1 = sgn_i(wv.y);
    const int s2 = sgn_i(wv.z);
    const int s3 = sgn_i(wv.w);
    const int packed = (s0 & 0xFF) | ((s1 & 0xFF) << 8) |
                       ((s2 & 0xFF) << 16) | (s3 << 24);

    const int tile = o >> 4, l = o & 15;
    const int kc   = k >> 6, kin = k & 63;
    const int j    = kin >> 5;
    const int half = (kin & 31) >> 4;
    const int b    = kin & 15;
    const size_t off =
        ((((size_t)tile * 16 + kc) * 2 + j) * 32 + (half * 16 + l)) * 16 + b;
    *(int*)(pw8 + off) = packed;

    red[threadIdx.x] = fabsf(wv.x) + fabsf(wv.y) + fabsf(wv.z) + fabsf(wv.w);
    __syncthreads();
    for (int s = 128; s > 0; s >>= 1) {
        if (threadIdx.x < s) red[threadIdx.x] += red[threadIdx.x + s];
        __syncthreads();
    }
    if (threadIdx.x == 0) atomicAdd(sum_abs, red[0]);
}

__global__ void __launch_bounds__(256)
hb_gemm_fused(const float* __restrict__ x, const float* __restrict__ bias,
              const char* __restrict__ pw8, const float* __restrict__ sum_abs,
              float* __restrict__ out) {
    __shared__ char atile[MB * LDS_STRIDE];

    const int tid   = threadIdx.x;
    const int wave  = tid >> 5;
    const int lane  = tid & 31;
    const int l     = lane & 15;
    const int half  = lane >> 4;
    const int tn0   = (blockIdx.x * 8 + wave) * 4;

    const long  xBase = (long)blockIdx.y * MB * DIN;
    const int   c4    = tid * 4;
    const float4 bv   = *(const float4*)(bias + c4);
    for (int s = 0; s < MB; ++s) {
        const float4 xv = *(const float4*)(x + xBase + (long)s * DIN + c4);
        const int q0 = sgn_i(xv.x + bv.x);
        const int q1 = sgn_i(xv.y + bv.y);
        const int q2 = sgn_i(xv.z + bv.z);
        const int q3 = sgn_i(xv.w + bv.w);
        *(int*)(atile + s * LDS_STRIDE + c4) =
            (q0 & 0xFF) | ((q1 & 0xFF) << 8) | ((q2 & 0xFF) << 16) | (q3 << 24);
    }
    __syncthreads();

    const char* bq    = pw8 + (size_t)tn0 * 16384 + (size_t)lane * 16;
    const char* abase = atile + l * LDS_STRIDE + half * 16;

    v8i acc[4][4];
#pragma unroll
    for (int mt = 0; mt < 4; ++mt)
#pragma unroll
        for (int t = 0; t < 4; ++t) acc[mt][t] = (v8i){};

    for (int kc = 0; kc < 16; ++kc) {
        v8i bb[4];
#pragma unroll
        for (int t = 0; t < 4; ++t) {
            const int4 q = *(const int4*)(bq + t * 16384 + kc * 1024);
            const int4 r = *(const int4*)(bq + t * 16384 + kc * 1024 + 512);
            bb[t] = (v8i){q.x, q.y, q.z, q.w, r.x, r.y, r.z, r.w};
        }
#pragma unroll
        for (int mt = 0; mt < 4; ++mt) {
            const char* ab = abase + mt * (16 * LDS_STRIDE) + kc * 64;
            const int2 p0 = *(const int2*)(ab);
            const int2 p1 = *(const int2*)(ab + 8);
            const int2 p2 = *(const int2*)(ab + 32);
            const int2 p3 = *(const int2*)(ab + 40);
            const v8i a = {p0.x, p0.y, p1.x, p1.y, p2.x, p2.y, p3.x, p3.y};
#pragma unroll
            for (int t = 0; t < 4; ++t)
                acc[mt][t] = __builtin_amdgcn_wmma_i32_16x16x64_iu8(
                    true, a, true, bb[t], acc[mt][t], false, false);
        }
    }

    const float scale = sum_abs[0] * (1.0f / ((float)DIN * (float)DOUT));
    const int colBase = tn0 * 16 + l;
#pragma unroll
    for (int mt = 0; mt < 4; ++mt) {
        const int rowBase = blockIdx.y * MB + mt * 16 + half * 8;
#pragma unroll
        for (int j = 0; j < 8; ++j) {
            float* orow = out + (long)(rowBase + j) * DOUT + colBase;
            orow[0]  = scale * (float)acc[mt][0][j];
            orow[16] = scale * (float)acc[mt][1][j];
            orow[32] = scale * (float)acc[mt][2][j];
            orow[48] = scale * (float)acc[mt][3][j];
        }
    }
}

__global__ void hb_abs_sum(const float* __restrict__ w,
                           float* __restrict__ sum_abs) {
    __shared__ float red[256];
    const long i0 = ((long)blockIdx.x * blockDim.x + threadIdx.x) * 4;
    const float4 wv = *(const float4*)(w + i0);
    red[threadIdx.x] = fabsf(wv.x) + fabsf(wv.y) + fabsf(wv.z) + fabsf(wv.w);
    __syncthreads();
    for (int s = 128; s > 0; s >>= 1) {
        if (threadIdx.x < s) red[threadIdx.x] += red[threadIdx.x + s];
        __syncthreads();
    }
    if (threadIdx.x == 0) atomicAdd(sum_abs, red[0]);
}

__global__ void hb_gemm_naive(const float* __restrict__ x,
                              const float* __restrict__ bias,
                              const float* __restrict__ w,
                              const float* __restrict__ sum_abs,
                              float* __restrict__ out) {
    const long idx = (long)blockIdx.x * blockDim.x + threadIdx.x;
    if (idx >= (long)NROWS * DOUT) return;
    const int n = (int)(idx / DOUT);
    const int o = (int)(idx % DOUT);
    int s = 0;
    for (int k = 0; k < DIN; ++k) {
        s += sgn_i(x[(long)n * DIN + k] + bias[k]) * sgn_i(w[(long)o * DIN + k]);
    }
    out[idx] = sum_abs[0] * (1.0f / ((float)DIN * (float)DOUT)) * (float)s;
}

extern "C" void kernel_launch(void* const* d_in, const int* in_sizes, int n_in,
                              void* d_out, int out_size, void* d_ws, size_t ws_size,
                              hipStream_t stream) {
    const float* x    = (const float*)d_in[0];
    const float* bias = (const float*)d_in[1];
    const float* w    = (const float*)d_in[2];
    float*       out  = (float*)d_out;

    char*  ws      = (char*)d_ws;
    float* sum_abs = (float*)ws;
    char*  pw8     = ws + 256;
    const size_t needed = 256 + (size_t)DOUT * DIN;

    hipMemsetAsync(sum_abs, 0, sizeof(float), stream);

    if (ws_size >= needed) {
        hb_quant_w_packed<<<((long)DOUT * DIN / 4) / 256, 256, 0, stream>>>(
            w, pw8, sum_abs);
        hb_gemm_fused<<<dim3(2, NROWS / MB), 256, 0, stream>>>(x, bias, pw8,
                                                               sum_abs, out);
    } else if (ws_size >= sizeof(float)) {
        hb_abs_sum<<<((long)DOUT * DIN / 4) / 256, 256, 0, stream>>>(w, sum_abs);
        hb_gemm_naive<<<((long)NROWS * DOUT + 255) / 256, 256, 0, stream>>>(
            x, bias, w, sum_abs, out);
    }
}
